// gcn_16681652978227
// MI455X (gfx1250) — matrix operand order repaired and checked by reading, not compiled, not run on hardware
//
#include <hip/hip_runtime.h>

// ---------------- sizes ----------------
#define NB    128      // batch N
#define NCH   32       // channels C
#define VDIM  207      // vertices V
#define LH    12       // length L
#define KPAD  224      // V padded to 7*32 for K of bf16 WMMA
#define WROWS 208      // V padded to 13*16 for M rows
#define COUT  64
#define CCAT  224      // 7*32 concat channels (exact multiple of 32)
#define LDP   232      // LDS pitch (c' dim padded)
#define NCOLT 24       // (c,l) column tiles: 32*12/16

typedef __attribute__((ext_vector_type(16))) __bf16         v16bf;
typedef __attribute__((ext_vector_type(8)))  float          v8f;
typedef __attribute__((ext_vector_type(16))) unsigned short v16u;
typedef __attribute__((ext_vector_type(8)))  unsigned short v8u;

__device__ __forceinline__ unsigned short f2bf(float f) {
  unsigned int u = __builtin_bit_cast(unsigned int, f);
  u += 0x7FFFu + ((u >> 16) & 1u);          // round-to-nearest-even
  return (unsigned short)(u >> 16);
}

// Build a 16-element bf16 fragment from two 16-byte chunks (global or LDS).
__device__ __forceinline__ v16bf frag2(const unsigned short* lo, const unsigned short* hi) {
  v8u a = *(const v8u*)lo;
  v8u b = *(const v8u*)hi;
  v16u r;
#pragma unroll
  for (int i = 0; i < 8; ++i) { r[i] = a[i]; r[i + 8] = b[i]; }
  return __builtin_bit_cast(v16bf, r);
}

// ------------- prep: M_g^T (A and A*A), transposed, padded, bf16 -------------
// Mg layout: [g][w(208)][v(224)], Mg[g][w][v] = M_g[v][w]; g: A1,A1^2,A2,A2^2,A3,A3^2
__global__ void prep_mats(const float* __restrict__ A1, const float* __restrict__ A2,
                          const float* __restrict__ A3, unsigned short* __restrict__ Mg) {
  int idx = blockIdx.x * blockDim.x + threadIdx.x;
  if (idx >= 6 * WROWS * KPAD) return;
  int v = idx % KPAD;
  int w = (idx / KPAD) % WROWS;
  int g = idx / (KPAD * WROWS);
  const float* A = (g < 2) ? A1 : ((g < 4) ? A2 : A3);
  float val = 0.f;
  if (v < VDIM && w < VDIM) {
    if ((g & 1) == 0) {
      val = A[v * VDIM + w];
    } else {
      float s = 0.f;
      for (int k = 0; k < VDIM; ++k) s += A[v * VDIM + k] * A[k * VDIM + w];
      val = s;
    }
  }
  Mg[idx] = f2bf(val);
}

// ------------- prep: x -> bf16, transposed to [nc][l(12)][v(224)] -------------
__global__ void prep_x(const float* __restrict__ x, unsigned short* __restrict__ xT) {
  int idx = blockIdx.x * blockDim.x + threadIdx.x;
  if (idx >= NB * NCH * LH * KPAD) return;
  int v  = idx % KPAD;
  int l  = (idx / KPAD) % LH;
  int nc = idx / (KPAD * LH);
  float val = (v < VDIM) ? x[(nc * VDIM + v) * LH + l] : 0.f;
  xT[idx] = f2bf(val);
}

// ------------- prep: W -> bf16 (already [64][224] row-major) -------------
__global__ void prep_w(const float* __restrict__ W, unsigned short* __restrict__ Wb) {
  int idx = blockIdx.x * blockDim.x + threadIdx.x;
  if (idx >= COUT * CCAT) return;
  Wb[idx] = f2bf(W[idx]);
}

// ------------- fused diffusion + 1x1-conv kernel -------------
// grid: (13 w-tiles, 128 n), block: 256 threads = 8 wave32
__launch_bounds__(256)
__global__ void gcn_fused(const unsigned short* __restrict__ Mg,
                          const unsigned short* __restrict__ xT,
                          const unsigned short* __restrict__ Wb,
                          const float* __restrict__ bias,
                          float* __restrict__ out) {
  extern __shared__ unsigned short h_lds[];   // [192 cols=(w,l)][LDP c'] bf16
  const int wt    = blockIdx.x;               // w-tile 0..12
  const int n     = blockIdx.y;               // 0..127
  const int tid   = threadIdx.x;
  const int lane  = tid & 31;
  const int wv    = tid >> 5;                 // wave 0..7
  const int lan16 = lane & 15;
  const int half  = lane >> 4;

  // ---- Phase 0: identity group (c' = 0..31) into h_lds[col][c'] ----
  for (int i = tid; i < 192 * NCH; i += 256) {
    int col = i >> 5;                         // col = w*12 + l
    int c   = i & 31;
    int w = col / 12, l = col % 12;
    int gw = wt * 16 + w;                     // <= 207 < KPAD (pad region is zero)
    h_lds[col * LDP + c] = xT[((n * NCH + c) * LH + l) * KPAD + gw];
  }

  // ---- Phase 1: diffusion terms, c' = 32..223 ----
  // N-dim packs (c,l): 384 valid columns = 24 tiles; B fragments register-cached
  // across all 6 diffusion matrices so xT is streamed from DRAM exactly once.
  for (int ct1 = wv; ct1 < NCOLT; ct1 += 8) {
    // this lane's (c,l) column for B fragments
    int colidx = ct1 * 16 + lan16;
    int bc = colidx / 12;
    int bl = colidx % 12;
    const unsigned short* brow = xT + ((n * NCH + bc) * LH + bl) * KPAD;

    // prefetch next column-tile's row for this lane
    if (ct1 + 8 < NCOLT) {
      int pcol = (ct1 + 8) * 16 + lan16;
      const unsigned short* prow =
          xT + ((n * NCH + pcol / 12) * LH + (pcol % 12)) * KPAD;
      __builtin_prefetch(prow, 0, 1);
    }

    v16bf bfr[7];
#pragma unroll
    for (int kb = 0; kb < 7; ++kb) {
      int k0 = kb * 32;
      bfr[kb] = frag2(brow + k0 + half * 8, brow + k0 + 16 + half * 8);
    }

#pragma unroll
    for (int g = 0; g < 6; ++g) {
      const unsigned short* arow = Mg + (g * WROWS + wt * 16 + lan16) * KPAD;
      v8f acc = {0.f, 0.f, 0.f, 0.f, 0.f, 0.f, 0.f, 0.f};
#pragma unroll
      for (int kb = 0; kb < 7; ++kb) {
        int k0 = kb * 32;
        v16bf a = frag2(arow + k0 + half * 8, arow + k0 + 16 + half * 8);
        acc = __builtin_amdgcn_wmma_f32_16x16x32_bf16(false, a, false, bfr[kb],
                                                      (short)0, acc, false, false);
      }
      // scatter tile into h_lds: rows w = r + half*8, column (bc,bl), c' = 32*(1+g)+bc
      int cp = 32 * (1 + g) + bc;
#pragma unroll
      for (int r = 0; r < 8; ++r) {
        int wloc = r + half * 8;
        h_lds[(wloc * 12 + bl) * LDP + cp] = f2bf(acc[r]);
      }
    }
  }

  __syncthreads();

  // ---- Phase 2: channel mix y[o, col] = sum_{c'} W[o,c'] * h[c', col] + b[o] ----
  for (int j = wv; j < 48; j += 8) {
    int ot = j / 12;                          // o-tile 0..3
    int ct = j % 12;                          // col-tile 0..11 (cols = (w,l))
    const unsigned short* wrow = Wb + (ot * 16 + lan16) * CCAT;
    const unsigned short* brow = &h_lds[(ct * 16 + lan16) * LDP];
    v8f acc = {0.f, 0.f, 0.f, 0.f, 0.f, 0.f, 0.f, 0.f};
#pragma unroll
    for (int kb = 0; kb < 7; ++kb) {
      int k0 = kb * 32;
      v16bf a = frag2(wrow + k0 + half * 8, wrow + k0 + 16 + half * 8);
      v16bf b = frag2(brow + k0 + half * 8, brow + k0 + 16 + half * 8);
      acc = __builtin_amdgcn_wmma_f32_16x16x32_bf16(false, a, false, b,
                                                    (short)0, acc, false, false);
    }
    int col = ct * 16 + lan16;
    int w = col / 12, l = col % 12;
    int gw = wt * 16 + w;
    if (gw < VDIM) {
#pragma unroll
      for (int r = 0; r < 8; ++r) {
        int o = ot * 16 + r + half * 8;
        out[((n * COUT + o) * VDIM + gw) * LH + l] = acc[r] + bias[o];
      }
    }
  }
}

// ---------------- launcher ----------------
extern "C" void kernel_launch(void* const* d_in, const int* in_sizes, int n_in,
                              void* d_out, int out_size, void* d_ws, size_t ws_size,
                              hipStream_t stream) {
  const float* x  = (const float*)d_in[0];
  const float* A1 = (const float*)d_in[1];
  const float* A2 = (const float*)d_in[2];
  const float* A3 = (const float*)d_in[3];
  const float* W  = (const float*)d_in[4];
  const float* b  = (const float*)d_in[5];
  float* out = (float*)d_out;

  // workspace layout (bytes):
  //   Mg: 6*208*224*2 = 559,104            @ 0
  //   xT: 4096*12*224*2 = 22,020,096       @ 561,152   (4 KB aligned)
  //   Wb: 64*224*2 = 28,672                @ 22,581,248
  char* ws = (char*)d_ws;
  unsigned short* Mg = (unsigned short*)(ws);
  unsigned short* xT = (unsigned short*)(ws + 561152);
  unsigned short* Wb = (unsigned short*)(ws + 22581248);
  (void)in_sizes; (void)n_in; (void)out_size; (void)ws_size;

  {
    int total = 6 * WROWS * KPAD;
    prep_mats<<<(total + 255) / 256, 256, 0, stream>>>(A1, A2, A3, Mg);
  }
  {
    int total = NB * NCH * LH * KPAD;
    prep_x<<<(total + 255) / 256, 256, 0, stream>>>(x, xT);
  }
  {
    int total = COUT * CCAT;
    prep_w<<<(total + 255) / 256, 256, 0, stream>>>(W, Wb);
  }
  {
    size_t lds_bytes = (size_t)192 * LDP * sizeof(unsigned short);  // 89,088 B
    gcn_fused<<<dim3(13, NB), 256, lds_bytes, stream>>>(Mg, xT, Wb, b, out);
  }
}
